// GigaGANTryOnGenerator_60129542637
// MI455X (gfx1250) — matrix operand order repaired and checked by reading, not compiled, not run on hardware
//
#include <hip/hip_runtime.h>
#include <math.h>

#define TPB 256

typedef unsigned short u16;
typedef __attribute__((ext_vector_type(16))) __bf16 v16bf;
typedef __attribute__((ext_vector_type(8)))  float  v8f;
typedef int v4i __attribute__((vector_size(16)));

#if __has_builtin(__builtin_amdgcn_global_load_async_to_lds_b128) && \
    __has_builtin(__builtin_amdgcn_s_wait_asynccnt)
#define HAVE_ASYNC 1
#else
#define HAVE_ASYNC 0
#endif

__device__ __forceinline__ u16 f2bf(float f) {
  union { float f; unsigned u; } v; v.f = f;
  unsigned u = v.u + (0x7FFFu + ((v.u >> 16) & 1u));  // round-to-nearest-even
  return (u16)(u >> 16);
}
__device__ __forceinline__ __bf16 bf_from_u16(u16 u) {
  union { u16 u; __bf16 b; } v; v.u = u; return v.b;
}

// ---------------------------------------------------------------------------
// Generic implicit-GEMM: D[b, n, m] = sum_k A[b, m, k] * Bw[b, k, n]
// Conv mode: A[m,k] gathered im2col from NCHW activations.
// Dense mode: A is a row-major [M,K] matrix per batch.
// Epilogue: +bias[n], +noise_w*noise[b,m], leaky-ReLU(0.2).
// Block tile 128(M) x 64(N); 8 waves, each wave: 16(M) x 64(N) strip =
// 4 accumulators reusing one A fragment -> 4 WMMAs per K-step of 32.
// B tile staged via CDNA5 async global->LDS (b128) when tile is full/aligned.
// ---------------------------------------------------------------------------
struct GemmP {
  const float* A; const u16* B; float* out;
  const float* bias; const float* noise; const float* noise_w;
  int denseA, leaky;
  int Cin, Hin, Win, KH, KW, stride, pad, Hout, Wout;
  int M, N, K;
  long aBatch, bBatch, oBatch;
};

__global__ __launch_bounds__(TPB) void gemm_conv_wmma(GemmP p) {
  __shared__ __attribute__((aligned(16))) u16 As[128][40];
  __shared__ __attribute__((aligned(16))) u16 Bs[32][72];
  const int b  = blockIdx.z;
  const int m0 = blockIdx.x * 128, n0 = blockIdx.y * 64;
  const float* A  = p.A + (long)b * p.aBatch;
  const u16*   Bw = p.B + (long)b * p.bBatch;
  const int tid  = threadIdx.x;
  const int wave = tid >> 5, lane = tid & 31;
  const int half = lane >> 4, l16 = lane & 15;
  const int khkw = p.KH * p.KW;
  v8f acc[4] = {{}, {}, {}, {}};
  for (int k0 = 0; k0 < p.K; k0 += 32) {
    // ---- stage A tile (128 M x 32 K), f32 -> bf16 on the fly ----
#pragma unroll 4
    for (int r = 0; r < 16; ++r) {
      int idx = tid + r * TPB;
      int ml = idx >> 5, kl = idx & 31;
      int m = m0 + ml, k = k0 + kl;
      float v = 0.f;
      if (m < p.M && k < p.K) {
        if (p.denseA) {
          v = A[(long)m * p.K + k];
        } else {
          int c = k / khkw, rs = k - c * khkw;
          int rr = rs / p.KW, ss = rs - rr * p.KW;
          int y = m / p.Wout, xo = m - y * p.Wout;
          int iy = y * p.stride - p.pad + rr;
          int ix = xo * p.stride - p.pad + ss;
          if (iy >= 0 && iy < p.Hin && ix >= 0 && ix < p.Win)
            v = A[((long)c * p.Hin + iy) * p.Win + ix];
        }
      }
      As[ml][kl] = f2bf(v);
    }
    // ---- stage B tile (32 K x 64 N), pre-packed bf16 [K,N] ----
    const bool fullB = (k0 + 32 <= p.K) && (n0 + 64 <= p.N) && ((p.N & 7) == 0);
#if HAVE_ASYNC
    if (fullB) {
      // 256 threads x 16B async chunks = the whole 4KB tile, no VGPR round-trip
      int row = tid >> 3, chunk = tid & 7;
      const u16* gsrc = Bw + (long)(k0 + row) * p.N + (n0 + chunk * 8);
      u16* ldst = &Bs[row][chunk * 8];
      __builtin_amdgcn_global_load_async_to_lds_b128(
          (__attribute__((address_space(1))) v4i*)(unsigned long long)gsrc,
          (__attribute__((address_space(3))) v4i*)(unsigned)(unsigned long long)ldst,
          0, 0);
    } else
#endif
    {
      for (int r = 0; r < 8; ++r) {
        int idx = tid + r * TPB;
        int kl = idx >> 6, nl = idx & 63;
        int k = k0 + kl, n = n0 + nl;
        Bs[kl][nl] = (k < p.K && n < p.N) ? Bw[(long)k * p.N + n] : (u16)0;
      }
    }
    if (k0 + 32 < p.K)  // CDNA5 global_prefetch of next weight tile
      __builtin_prefetch(Bw + (long)(k0 + 32) * p.N + n0 + lane, 0, 1);
#if HAVE_ASYNC
    if (fullB) __builtin_amdgcn_s_wait_asynccnt(0);
#endif
    __syncthreads();
    // ---- fragments per the CDNA5 16-bit WMMA VGPR layouts ----
    v16bf af;
    const int arow = wave * 16 + l16;
#pragma unroll
    for (int e = 0; e < 16; ++e) {
      int j = e >> 1;
      int kk = ((j < 4) ? (8 * half + 2 * j) : (16 + 8 * half + 2 * (j - 4))) + (e & 1);
      af[e] = bf_from_u16(As[arow][kk]);
    }
    v16bf bfv[4];
#pragma unroll
    for (int jn = 0; jn < 4; ++jn) {
      const int bcol = jn * 16 + l16;
#pragma unroll
      for (int e = 0; e < 16; ++e)
        bfv[jn][e] = bf_from_u16(Bs[8 * half + e + ((e >= 8) ? 8 : 0)][bcol]);
    }
#pragma unroll
    for (int jn = 0; jn < 4; ++jn)
      acc[jn] = __builtin_amdgcn_wmma_f32_16x16x32_bf16(
          false, af, false, bfv[jn], (short)0, acc[jn], false, false);
    __syncthreads();
  }
  const float nw = p.noise_w ? p.noise_w[0] : 0.f;
#pragma unroll
  for (int jn = 0; jn < 4; ++jn) {
#pragma unroll
    for (int r = 0; r < 8; ++r) {
      int m = m0 + wave * 16 + half * 8 + r;
      int n = n0 + jn * 16 + l16;
      if (m < p.M && n < p.N) {
        float v = acc[jn][r];
        if (p.bias) v += p.bias[n];
        if (p.noise) {
          int y = m / p.Wout, xo = m - y * p.Wout;
          v += nw * p.noise[((long)b * p.Hout + y) * p.Wout + xo];
        }
        if (p.leaky) v = v > 0.f ? v : 0.2f * v;
        p.out[(long)b * p.oBatch + (long)n * p.M + m] = v;
      }
    }
  }
}

// ---------------------------------------------------------------------------
// Style-blend + modulate + demodulate -> packed bf16 weights [b, K, O]
// one block per (b, o); K values staged in dynamic LDS for the two passes
// ---------------------------------------------------------------------------
__global__ __launch_bounds__(TPB) void blend_modw(
    const float* W, const float* kattn, const float* smod, u16* out,
    int nk, int O, int C, int KH, int KW, float base, int demod) {
  const int o = blockIdx.x % O, b = blockIdx.x / O;
  const int K = C * KH * KW;
  extern __shared__ float sm[];
  __shared__ float red[8];
  __shared__ float tot;
  float a[4];
  for (int t = 0; t < 4; ++t) a[t] = (t < nk) ? kattn[b * nk + t] : 0.f;
  const long oslice = (long)o * K;
  const long kstride = (long)O * K;
  const int khkw = KH * KW;
  float ss = 0.f;
  for (int t = threadIdx.x; t < K; t += blockDim.x) {
    int c = t / khkw;
    float v = a[0] * W[oslice + t];
    if (nk > 1) v += a[1] * W[kstride + oslice + t];
    if (nk > 2) v += a[2] * W[2 * kstride + oslice + t];
    if (nk > 3) v += a[3] * W[3 * kstride + oslice + t];
    v *= base * smod[(long)b * C + c];
    sm[t] = v;
    ss += v * v;
  }
  for (int m = 16; m; m >>= 1) ss += __shfl_xor(ss, m, 32);
  if ((threadIdx.x & 31) == 0) red[threadIdx.x >> 5] = ss;
  __syncthreads();
  if (threadIdx.x == 0) {
    float s = 0.f;
    for (int i = 0; i < 8; ++i) s += red[i];
    tot = s;
  }
  __syncthreads();
  const float f = demod ? rsqrtf(tot + 1e-8f) : 1.f;
  for (int t = threadIdx.x; t < K; t += blockDim.x)
    out[((long)b * K + t) * O + o] = f2bf(sm[t] * f);
}

// one wave (32 lanes) per (b, o) dot product
__global__ __launch_bounds__(TPB) void eqlinear_k(
    const float* x, const float* w, const float* bias, float* out,
    int Bn, int O, int Cin, float scale, int leaky) {
  int gw = (int)((blockIdx.x * blockDim.x + threadIdx.x) >> 5);
  int lane = threadIdx.x & 31;
  if (gw >= Bn * O) return;
  int b = gw / O, o = gw - b * O;
  float acc = 0.f;
  for (int c = lane; c < Cin; c += 32)
    acc += x[(long)b * Cin + c] * w[(long)o * Cin + c];
  for (int m = 16; m; m >>= 1) acc += __shfl_xor(acc, m, 32);
  if (lane == 0) {
    float v = acc * scale + bias[o];
    if (leaky) v = v > 0.f ? v : 0.2f * v;
    out[(long)b * O + o] = v;
  }
}

__global__ __launch_bounds__(TPB) void mean_hw_k(const float* x, float* out, long HW) {
  const float* p = x + (long)blockIdx.x * HW;
  __shared__ float red[8];
  float s = 0.f;
  for (long i = threadIdx.x; i < HW; i += blockDim.x) s += p[i];
  for (int m = 16; m; m >>= 1) s += __shfl_xor(s, m, 32);
  if ((threadIdx.x & 31) == 0) red[threadIdx.x >> 5] = s;
  __syncthreads();
  if (threadIdx.x == 0) {
    float t = 0.f;
    for (int i = 0; i < 8; ++i) t += red[i];
    out[blockIdx.x] = t / (float)HW;
  }
}

__global__ __launch_bounds__(TPB) void softmax_rows_k(float* data, int cols) {
  float* d = data + (long)blockIdx.x * cols;
  __shared__ float red[8];
  __shared__ float bval;
  int tid = threadIdx.x;
  float mx = -3.0e38f;
  for (int c = tid; c < cols; c += blockDim.x) mx = fmaxf(mx, d[c]);
  for (int m = 16; m; m >>= 1) mx = fmaxf(mx, __shfl_xor(mx, m, 32));
  if ((tid & 31) == 0) red[tid >> 5] = mx;
  __syncthreads();
  if (tid == 0) {
    float v = -3.0e38f;
    for (int i = 0; i < 8; ++i) v = fmaxf(v, red[i]);
    bval = v;
  }
  __syncthreads();
  float rowmax = bval;
  float s = 0.f;
  for (int c = tid; c < cols; c += blockDim.x) {
    float e = __expf(d[c] - rowmax);
    d[c] = e;
    s += e;
  }
  for (int m = 16; m; m >>= 1) s += __shfl_xor(s, m, 32);
  if ((tid & 31) == 0) red[tid >> 5] = s;
  __syncthreads();
  if (tid == 0) {
    float v = 0.f;
    for (int i = 0; i < 8; ++i) v += red[i];
    bval = v;
  }
  __syncthreads();
  float inv = 1.f / bval;
  for (int c = tid; c < cols; c += blockDim.x) d[c] *= inv;
}

// pack static conv weights [O, C, kh, kw] -> bf16 [K, O], K = C*kh*kw
__global__ void pack_w_k(const float* w, u16* dst, int O, long K) {
  long i = (long)blockIdx.x * blockDim.x + threadIdx.x;
  long tot = (long)O * K;
  if (i >= tot) return;
  long k = i / O; int o = (int)(i - k * O);
  dst[i] = f2bf(w[(long)o * K + k]);
}

// pack f32 matrix -> bf16 [K, N], optional transpose of [N, K] source, *scale
__global__ void pack_dense_k(const float* src, u16* dst, int Bn, int K, int N,
                             long sB, long dB, int transp, float scale) {
  long i = (long)blockIdx.x * blockDim.x + threadIdx.x;
  long tot = (long)Bn * K * N;
  if (i >= tot) return;
  int n = (int)(i % N); long t = i / N; int k = (int)(t % K); int b = (int)(t / K);
  float v = transp ? src[(long)b * sB + (long)n * K + k]
                   : src[(long)b * sB + (long)k * N + n];
  dst[(long)b * dB + (long)k * N + n] = f2bf(v * scale);
}

// bilinear, half-pixel centers (align_corners=False)
__global__ void bilinear_k(const float* in, float* out, long BC,
                           int Hi, int Wi, int Ho, int Wo) {
  long i = (long)blockIdx.x * blockDim.x + threadIdx.x;
  long tot = BC * Ho * Wo;
  if (i >= tot) return;
  int x = (int)(i % Wo); long t = i / Wo; int y = (int)(t % Ho); long bc = t / Ho;
  float sy = (float)Hi / (float)Ho, sx = (float)Wi / (float)Wo;
  float fy = ((float)y + 0.5f) * sy - 0.5f;
  float fx = ((float)x + 0.5f) * sx - 0.5f;
  int y0 = (int)floorf(fy), x0 = (int)floorf(fx);
  float wy = fy - (float)y0, wx = fx - (float)x0;
  int y1 = y0 + 1, x1 = x0 + 1;
  y0 = min(max(y0, 0), Hi - 1); y1 = min(max(y1, 0), Hi - 1);
  x0 = min(max(x0, 0), Wi - 1); x1 = min(max(x1, 0), Wi - 1);
  const float* p = in + bc * (long)Hi * Wi;
  float v00 = p[(long)y0 * Wi + x0], v01 = p[(long)y0 * Wi + x1];
  float v10 = p[(long)y1 * Wi + x0], v11 = p[(long)y1 * Wi + x1];
  out[i] = (1.f - wy) * ((1.f - wx) * v00 + wx * v01)
         + wy * ((1.f - wx) * v10 + wx * v11);
}

__global__ void copy_chan_k(const float* src, float* dst, int Bn, int Cs,
                            long HW, int Cd, int off) {
  long i = (long)blockIdx.x * blockDim.x + threadIdx.x;
  long tot = (long)Bn * Cs * HW;
  if (i >= tot) return;
  long pos = i % HW; long t = i / HW; int c = (int)(t % Cs); int b = (int)(t / Cs);
  dst[((long)b * Cd + off + c) * HW + pos] = src[i];
}

__global__ void residual_k(const float* x, const float* a, const float* gamma,
                           float* out, long n) {
  long i = (long)blockIdx.x * blockDim.x + threadIdx.x;
  if (i >= n) return;
  out[i] = x[i] + gamma[0] * a[i];
}

__global__ void add2_k(const float* a, const float* b, float* out, long n) {
  long i = (long)blockIdx.x * blockDim.x + threadIdx.x;
  if (i >= n) return;
  out[i] = a[i] + b[i];
}

__global__ void tanh_k(const float* a, float* out, long n) {
  long i = (long)blockIdx.x * blockDim.x + threadIdx.x;
  if (i >= n) return;
  out[i] = tanhf(a[i]);
}

// ---------------------------------------------------------------------------
static inline long cdivl(long a, long b) { return (a + b - 1) / b; }
static inline dim3 g1(long n) { return dim3((unsigned)cdivl(n, TPB)); }

static void launchGemm(hipStream_t st, const float* A, const u16* Bw, float* out,
                       const float* bias, const float* noise, const float* noise_w,
                       int batches, int denseA,
                       int Cin, int Hin, int Win, int KH, int KW, int stride, int pad,
                       int Hout, int Wout, int N, int Mdense, int Kdense,
                       long aBatch, long bBatch, int leaky) {
  GemmP p{};
  p.A = A; p.B = Bw; p.out = out; p.bias = bias; p.noise = noise; p.noise_w = noise_w;
  p.denseA = denseA; p.leaky = leaky;
  p.Cin = Cin; p.Hin = Hin; p.Win = Win; p.KH = KH; p.KW = KW;
  p.stride = stride; p.pad = pad; p.Hout = Hout; p.Wout = Wout;
  p.M = denseA ? Mdense : Hout * Wout;
  p.K = denseA ? Kdense : Cin * KH * KW;
  p.N = N;
  p.aBatch = aBatch; p.bBatch = bBatch; p.oBatch = (long)p.N * p.M;
  dim3 grid((unsigned)cdivl(p.M, 128), (unsigned)cdivl(p.N, 64), batches);
  gemm_conv_wmma<<<grid, dim3(TPB), 0, st>>>(p);
}

// ---------------------------------------------------------------------------
extern "C" void kernel_launch(void* const* d_in, const int* in_sizes, int n_in,
                              void* d_out, int out_size, void* d_ws, size_t ws_size,
                              hipStream_t stream) {
  (void)in_sizes; (void)n_in; (void)out_size; (void)ws_size;
  const int ENC_CH[8] = {64, 128, 256, 512, 512, 512, 512, 512};
  const int SYN_CH[8] = {512, 512, 512, 512, 256, 128, 64, 32};
  const float INV_SQRT512 = 1.0f / sqrtf(512.f);

  // Assumed leaf order (jax pytree: dict keys sorted at every level, lists in order):
  //   0..15   noise[0..15]
  //   16..22  dec_attn {gamma, k.b, k.w, q.b, q.w, v.b, v.w}
  //   23..29  enc.attn {gamma, k.b, k.w, q.b, q.w, v.b, v.w}
  //   30..57  enc.blocks[7] {c1.b, c1.w, c2.b, c2.w}
  //   58,59   enc.initial {b, w}
  //   60..75  enc.style_proj[8] {b, w}
  //   76,77   seed {b, w}
  //   78..93  skips[8] {b, w}
  //   94..189 styled[16] {attn.b, attn.w, mod.b, mod.w, weight, noise_w}
  //   190..237 torgb[8] {bias, attn.b, attn.w, mod.b, mod.w, weight}
  //   238     x
  auto F = [&](int i) -> const float* { return (const float*)d_in[i]; };
  const float* x_in = F(238);
  auto NOISE = [&](int i) { return F(i); };
  auto STY = [&](int j, int l) { return F(94 + 6 * j + l); };
  auto TRG = [&](int i, int l) { return F(190 + 6 * i + l); };

  size_t off = 0;
  auto allocF = [&](long elems) -> float* {
    float* p = (float*)((char*)d_ws + off);
    off += ((size_t)elems * 4 + 255) & ~(size_t)255;
    return p;
  };
  auto allocU = [&](long elems) -> u16* {
    u16* p = (u16*)((char*)d_ws + off);
    off += ((size_t)elems * 2 + 255) & ~(size_t)255;
    return p;
  };

  // persistent encoder features
  float* feats[8];
  for (int i = 0; i < 8; ++i) {
    int s = 512 >> i;
    feats[i] = allocF(2L * ENC_CH[i] * s * s);
  }
  float* attnY = allocF(2L * 512 * 32 * 32);
  // scratch pools
  float* P0 = allocF(2L * 96 * 512 * 512);  // upsampled cat / encoder temp
  float* P1 = allocF(2L * 32 * 512 * 512);  // feat ping
  float* P2 = allocF(2L * 32 * 512 * 512);  // feat pong
  float* P3 = allocF(2L * 32 * 512 * 512);  // skip / attn-out
  float* P4 = allocF(2L * 96 * 256 * 256);  // concat at pre-upsample res
  float* R0 = allocF(2L * 3 * 512 * 512);
  float* R1 = allocF(2L * 3 * 512 * 512);
  float* R2 = allocF(2L * 3 * 512 * 512);
  u16* packW   = allocU(4608L * 512);       // static conv weights, packed
  u16* packMod = allocU(2L * 9216 * 512);   // per-batch modulated weights
  float* stylesBuf = allocF(8L * 2 * 512);
  float* meanb = allocF(2L * 1024);
  float* kattb = allocF(8);
  float* modsb = allocF(2L * 1024);
  // attention scratch (32x32, HW=1024, mid=64, ch=512)
  const int AH = 32, AW = 32, AHW = 1024, AMID = 64, ACH = 512;
  float* qbuf = allocF(2L * AMID * AHW);
  float* kbuf = allocF(2L * AMID * AHW);
  float* vbuf = allocF(2L * ACH * AHW);
  float* scores = allocF(2L * AHW * AHW);
  u16* qpack = allocU(2L * AMID * AHW);
  u16* vpack = allocU(2L * AHW * ACH);

  auto pack_static = [&](const float* w, int O, long K) {
    pack_w_k<<<g1((long)O * K), dim3(TPB), 0, stream>>>(w, packW, O, K);
  };

  auto self_attn = [&](const float* xin, float* yout, float* attnout,
                       int gI, int kbI, int kwI, int qbI, int qwI, int vbI, int vwI) {
    pack_static(F(qwI), AMID, (long)ACH);
    launchGemm(stream, xin, packW, qbuf, F(qbI), nullptr, nullptr, 2, 0,
               ACH, AH, AW, 1, 1, 1, 0, AH, AW, AMID, 0, 0, (long)ACH * AHW, 0, 0);
    pack_static(F(kwI), AMID, (long)ACH);
    launchGemm(stream, xin, packW, kbuf, F(kbI), nullptr, nullptr, 2, 0,
               ACH, AH, AW, 1, 1, 1, 0, AH, AW, AMID, 0, 0, (long)ACH * AHW, 0, 0);
    pack_static(F(vwI), ACH, (long)ACH);
    launchGemm(stream, xin, packW, vbuf, F(vbI), nullptr, nullptr, 2, 0,
               ACH, AH, AW, 1, 1, 1, 0, AH, AW, ACH, 0, 0, (long)ACH * AHW, 0, 0);
    // scores[b,i,j] = sum_d k[d,j] * q[d,i] / sqrt(mid)   (softmax over j)
    pack_dense_k<<<g1(2L * AMID * AHW), dim3(TPB), 0, stream>>>(
        qbuf, qpack, 2, AMID, AHW, (long)AMID * AHW, (long)AMID * AHW, 0,
        1.0f / sqrtf((float)AMID));
    launchGemm(stream, kbuf, qpack, scores, nullptr, nullptr, nullptr, 2, 0,
               AMID, AH, AW, 1, 1, 1, 0, AH, AW, AHW, 0, 0,
               (long)AMID * AHW, (long)AMID * AHW, 0);
    softmax_rows_k<<<dim3(2 * AHW), dim3(TPB), 0, stream>>>(scores, AHW);
    // out[b,c,i] = sum_j attn[i,j] * v[c,j]
    pack_dense_k<<<g1(2L * AHW * ACH), dim3(TPB), 0, stream>>>(
        vbuf, vpack, 2, AHW, ACH, (long)ACH * AHW, (long)AHW * ACH, 1, 1.f);
    launchGemm(stream, scores, vpack, attnout, nullptr, nullptr, nullptr, 2, 1,
               0, AH, AW, 1, 1, 1, 0, AH, AW, ACH, AHW, AHW,
               (long)AHW * AHW, (long)AHW * ACH, 0);
    long n = 2L * ACH * AHW;
    residual_k<<<g1(n), dim3(TPB), 0, stream>>>(xin, attnout, F(gI), yout, n);
  };

  auto style_of = [&](int si, const float* feat, int ch, long hw, int pbI, int pwI) {
    mean_hw_k<<<dim3(2 * ch), dim3(TPB), 0, stream>>>(feat, meanb, hw);
    eqlinear_k<<<g1(2L * 512 * 32), dim3(TPB), 0, stream>>>(
        meanb, F(pwI), F(pbI), stylesBuf + (long)si * 1024, 2, 512, ch,
        1.0f / sqrtf((float)ch), 1);
  };

  // ---------------- encoder ----------------
  pack_static(F(59), 64, 54L);
  launchGemm(stream, x_in, packW, feats[0], F(58), nullptr, nullptr, 2, 0,
             6, 512, 512, 3, 3, 1, 1, 512, 512, 64, 0, 0, 6L * 512 * 512, 0, 1);
  style_of(0, feats[0], 64, 512L * 512, 60, 61);
  int prevc = 64, hs = 512;
  for (int blk = 0; blk < 7; ++blk) {
    int ch = ENC_CH[blk + 1];
    int ho = hs / 2;
    pack_static(F(30 + 4 * blk + 1), ch, (long)prevc * 9);
    launchGemm(stream, feats[blk], packW, P0, F(30 + 4 * blk + 0), nullptr, nullptr, 2, 0,
               prevc, hs, hs, 3, 3, 2, 1, ho, ho, ch, 0, 0, (long)prevc * hs * hs, 0, 1);
    pack_static(F(30 + 4 * blk + 3), ch, (long)ch * 9);
    launchGemm(stream, P0, packW, feats[blk + 1], F(30 + 4 * blk + 2), nullptr, nullptr, 2, 0,
               ch, ho, ho, 3, 3, 1, 1, ho, ho, ch, 0, 0, (long)ch * ho * ho, 0, 1);
    if (blk == 3) {  // self-attention at 32x32
      self_attn(feats[4], attnY, P3, 23, 24, 25, 26, 27, 28, 29);
      feats[4] = attnY;
    }
    style_of(blk + 1, feats[blk + 1], ch, (long)ho * ho, 60 + 2 * (blk + 1),
             61 + 2 * (blk + 1));
    prevc = ch; hs = ho;
  }

  // ---------------- decoder ----------------
  pack_static(F(77), 512, 4608L);
  launchGemm(stream, feats[7], packW, P1, F(76), nullptr, nullptr, 2, 0,
             512, 4, 4, 3, 3, 1, 1, 4, 4, 512, 0, 0, 512L * 16, 0, 1);
  float* fcur = P1; float* foth = P2;
  int prev = 512, fs = 4;

  for (int i = 0; i < 8; ++i) {
    const int ch = SYN_CH[i];
    const float* conv_in; int in_ch;
    if (i == 0) {
      conv_in = fcur; in_ch = prev;
    } else {
      const int sskip = 4 << i;
      const int ech = ENC_CH[7 - i];
      pack_static(F(78 + 2 * i + 1), ch, (long)ech);
      launchGemm(stream, feats[7 - i], packW, P3, F(78 + 2 * i), nullptr, nullptr, 2, 0,
                 ech, sskip, sskip, 1, 1, 1, 0, sskip, sskip, ch, 0, 0,
                 (long)ech * sskip * sskip, 0, 0);
      bilinear_k<<<g1(2L * ch * fs * fs), dim3(TPB), 0, stream>>>(
          P3, P0, 2L * ch, sskip, sskip, fs, fs);
      in_ch = prev + ch;
      copy_chan_k<<<g1(2L * prev * fs * fs), dim3(TPB), 0, stream>>>(
          fcur, P4, 2, prev, (long)fs * fs, in_ch, 0);
      copy_chan_k<<<g1(2L * ch * fs * fs), dim3(TPB), 0, stream>>>(
          P0, P4, 2, ch, (long)fs * fs, in_ch, prev);
      bilinear_k<<<g1(2L * in_ch * (2 * fs) * (2 * fs)), dim3(TPB), 0, stream>>>(
          P4, P0, 2L * in_ch, fs, fs, 2 * fs, 2 * fs);
      fs *= 2;
      conv_in = P0;
    }

    const float* st1 = stylesBuf + (long)((3 * i) % 8) * 1024;
    const float* st2 = stylesBuf + (long)((3 * i + 1) % 8) * 1024;
    const float* st3 = stylesBuf + (long)((3 * i + 2) % 8) * 1024;

    auto styled_do = [&](int j, const float* style, const float* src, float* dst,
                         int Cin_, int O_, const float* noisePtr) {
      eqlinear_k<<<g1(2L * 4 * 32), dim3(TPB), 0, stream>>>(
          style, STY(j, 1), STY(j, 0), kattb, 2, 4, 512, INV_SQRT512, 0);
      softmax_rows_k<<<dim3(2), dim3(TPB), 0, stream>>>(kattb, 4);
      eqlinear_k<<<g1(2L * Cin_ * 32), dim3(TPB), 0, stream>>>(
          style, STY(j, 3), STY(j, 2), modsb, 2, Cin_, 512, INV_SQRT512, 0);
      const int Km = Cin_ * 9;
      blend_modw<<<dim3(2 * O_), dim3(TPB), (size_t)Km * 4, stream>>>(
          STY(j, 4), kattb, modsb, packMod, 4, O_, Cin_, 3, 3,
          1.0f / sqrtf((float)Km), 1);
      launchGemm(stream, src, packMod, dst, nullptr, noisePtr, STY(j, 5), 2, 0,
                 Cin_, fs, fs, 3, 3, 1, 1, fs, fs, O_, 0, 0,
                 (long)Cin_ * fs * fs, (long)Km * O_, 1);
    };

    styled_do(2 * i, st1, conv_in, foth, in_ch, ch, NOISE(2 * i));
    styled_do(2 * i + 1, st2, foth, fcur, ch, ch, NOISE(2 * i + 1));

    if (i == 3) {  // decoder self-attention at 32x32
      self_attn(fcur, foth, P3, 16, 17, 18, 19, 20, 21, 22);
      float* t = fcur; fcur = foth; foth = t;
    }

    // toRGB (modulated 1x1, no demod) + progressive sum
    eqlinear_k<<<g1(2L * 1 * 32), dim3(TPB), 0, stream>>>(
        st3, TRG(i, 2), TRG(i, 1), kattb, 2, 1, 512, INV_SQRT512, 0);
    softmax_rows_k<<<dim3(2), dim3(TPB), 0, stream>>>(kattb, 1);
    eqlinear_k<<<g1(2L * ch * 32), dim3(TPB), 0, stream>>>(
        st3, TRG(i, 4), TRG(i, 3), modsb, 2, ch, 512, INV_SQRT512, 0);
    blend_modw<<<dim3(2 * 3), dim3(TPB), (size_t)ch * 4, stream>>>(
        TRG(i, 5), kattb, modsb, packMod, 1, 3, ch, 1, 1,
        1.0f / sqrtf((float)ch), 0);
    float* rgbdst = (i == 0) ? R1 : R0;
    launchGemm(stream, fcur, packMod, rgbdst, TRG(i, 0), nullptr, nullptr, 2, 0,
               ch, fs, fs, 1, 1, 1, 0, fs, fs, 3, 0, 0,
               (long)ch * fs * fs, (long)ch * 3, 0);
    if (i > 0) {
      long n = 2L * 3 * fs * fs;
      bilinear_k<<<g1(n), dim3(TPB), 0, stream>>>(R1, R2, 2L * 3, fs / 2, fs / 2, fs, fs);
      add2_k<<<g1(n), dim3(TPB), 0, stream>>>(R2, R0, R1, n);
    }
    prev = ch;
  }

  long nout = 2L * 3 * 512 * 512;
  tanh_k<<<g1(nout), dim3(TPB), 0, stream>>>(R1, (float*)d_out, nout);
}
